// MambaResidualDiffusion_68032281968703
// MI455X (gfx1250) — matrix operand order repaired and checked by reading, not compiled, not run on hardware
//
#include <hip/hip_runtime.h>
#include <hip/hip_bf16.h>
#include <math.h>

typedef __attribute__((ext_vector_type(16))) __bf16 v16bf;
typedef __attribute__((ext_vector_type(8)))  float  v8f;
typedef __attribute__((ext_vector_type(4)))  float  v4f;

#define B_  2
#define T_  2048
#define D_  128
#define H_  512
#define NL_ 3
#define E_  1024
#define NS_ 16      // SSM state dim
#define R_  32
#define KC_ 4       // mamba depthwise conv taps
#define M_  (B_*T_) // 4096 rows

// ---------------------------------------------------------------------------
// Weight pack: out[n*K + k] = (bf16) in[off + n*sn + k*sk]   (transposed B)
// ---------------------------------------------------------------------------
__global__ void k_pack(const float* __restrict__ in, __bf16* __restrict__ out,
                       long long off, long long sn, long long sk, int N, int K) {
  long long i = (long long)blockIdx.x * blockDim.x + threadIdx.x;
  if (i >= (long long)N * K) return;
  int n = (int)(i / K), k = (int)(i % K);
  out[i] = (__bf16)in[off + (long long)n * sn + (long long)k * sk];
}

__global__ void k_zero(float* __restrict__ p, int n) {
  int i = blockIdx.x * blockDim.x + threadIdx.x;
  if (i < n) p[i] = 0.0f;
}

// ---------------------------------------------------------------------------
// WMMA GEMM: C[M,N] (+=) act( A[M,K](f32, row remapped by shift within T-chunks)
//                             @ BT[N,K](bf16, pre-transposed)  + bias )
// One wave computes a 16x64 C strip (4 x 16x16 tiles, A reused 4x).
// Invalid A rows (OOB / conv padding) read from a zeroed scratch row -> no
// divergent loads in the K loop. ACT: 0=none 1=gelu(exact) 2=softplus
// ---------------------------------------------------------------------------
template<int ACT, bool ACC>
__global__ __launch_bounds__(32)
void wmma_gemm(const float* __restrict__ A, const __bf16* __restrict__ BT,
               const float* __restrict__ bias, float* __restrict__ C,
               const float* __restrict__ zrow,
               int M, int N, int K, int lda, int ldc, int shift, int Tper) {
  const int lane = threadIdx.x;
  const int l16 = lane & 15, half = lane >> 4;
  const int m0 = blockIdx.y * 16, n0 = blockIdx.x * 64;

  // A-fragment row for this lane (with conv shift + per-sequence masking)
  int mrow = m0 + l16;
  int bidx = mrow / Tper, t = mrow % Tper;
  int t2 = t + shift;
  bool mvalid = (mrow < M) && (t2 >= 0) && (t2 < Tper);
  const float* arow = mvalid ? (A + ((long long)bidx * Tper + t2) * lda) : zrow;

  const __bf16* brow = BT + (long long)(n0 + l16) * K + half * 16;
  const long long bstep = (long long)16 * K;   // next 16-col tile

  v8f acc0, acc1, acc2, acc3;
#pragma unroll
  for (int v = 0; v < 8; ++v) { acc0[v] = 0.f; acc1[v] = 0.f; acc2[v] = 0.f; acc3[v] = 0.f; }

  for (int k0 = 0; k0 < K; k0 += 32) {
    const float* ap = arow + k0 + half * 16;
    v4f f0 = *(const v4f*)(ap);
    v4f f1 = *(const v4f*)(ap + 4);
    v4f f2 = *(const v4f*)(ap + 8);
    v4f f3 = *(const v4f*)(ap + 12);
    v16bf a;
#pragma unroll
    for (int j = 0; j < 4; ++j) {
      a[j]      = (__bf16)f0[j];
      a[4 + j]  = (__bf16)f1[j];
      a[8 + j]  = (__bf16)f2[j];
      a[12 + j] = (__bf16)f3[j];
    }
    v16bf b0 = *(const v16bf*)(brow + k0);
    v16bf b1 = *(const v16bf*)(brow + bstep + k0);
    v16bf b2 = *(const v16bf*)(brow + 2 * bstep + k0);
    v16bf b3 = *(const v16bf*)(brow + 3 * bstep + k0);
    acc0 = __builtin_amdgcn_wmma_f32_16x16x32_bf16(false, a, false, b0, (short)0, acc0, false, false);
    acc1 = __builtin_amdgcn_wmma_f32_16x16x32_bf16(false, a, false, b1, (short)0, acc1, false, false);
    acc2 = __builtin_amdgcn_wmma_f32_16x16x32_bf16(false, a, false, b2, (short)0, acc2, false, false);
    acc3 = __builtin_amdgcn_wmma_f32_16x16x32_bf16(false, a, false, b3, (short)0, acc3, false, false);
  }

#pragma unroll
  for (int j = 0; j < 4; ++j) {
    const v8f acc = (j == 0) ? acc0 : (j == 1) ? acc1 : (j == 2) ? acc2 : acc3;
    const int col = n0 + j * 16 + l16;
    const float bv = bias ? bias[col] : 0.0f;
#pragma unroll
    for (int v = 0; v < 8; ++v) {
      int row = m0 + half * 8 + v;
      if (row < M) {
        float val = acc[v] + bv;
        if (ACT == 1)      val = 0.5f * val * (1.0f + erff(val * 0.70710678f));
        else if (ACT == 2) val = (val > 20.0f) ? val : log1pf(expf(val));
        long long idx = (long long)row * ldc + col;
        if (ACC) C[idx] += val; else C[idx] = val;
      }
    }
  }
}

// ---------------------------------------------------------------------------
// Elementwise / small kernels
// ---------------------------------------------------------------------------
__global__ void k_prep(const float* __restrict__ xt, const float* __restrict__ last,
                       const float* __restrict__ fe, float* __restrict__ xin) {
  long long i = (long long)blockIdx.x * blockDim.x + threadIdx.x;
  if (i >= (long long)B_ * T_ * D_) return;
  int d = (int)(i % D_);
  int b = (int)(i / ((long long)T_ * D_));
  xin[i] = xt[i] - last[b * D_ + d] + fe[d];
}

__global__ void k_temb(const int* __restrict__ tt, const float* __restrict__ w1,
                       const float* __restrict__ b1, const float* __restrict__ w2,
                       const float* __restrict__ b2, float* __restrict__ temb) {
  __shared__ float s[H_];
  int b = blockIdx.x, h = threadIdx.x;
  float tf = (float)tt[b] * 0.001f;
  float pre = tf * w1[h] + b1[h];
  s[h] = pre / (1.0f + expf(-pre));
  __syncthreads();
  float acc = b2[h];
  for (int j = 0; j < H_; ++j) acc += s[j] * w2[j * H_ + h];
  temb[b * H_ + h] = acc;
}

__global__ void k_addtemb(float* __restrict__ x, const float* __restrict__ temb) {
  long long i = (long long)blockIdx.x * blockDim.x + threadIdx.x;
  if (i >= (long long)M_ * H_) return;
  int h = (int)(i % H_);
  int b = (int)(i / ((long long)T_ * H_));
  x[i] += temb[b * H_ + h];
}

__global__ void k_rmsnorm(const float* __restrict__ x, const float* __restrict__ w,
                          float* __restrict__ xn) {
  __shared__ float red[128];
  long long row = blockIdx.x;
  const float* xr = x + row * H_;
  float s = 0.0f;
  for (int j = threadIdx.x; j < H_; j += 128) { float v = xr[j]; s += v * v; }
  red[threadIdx.x] = s; __syncthreads();
  for (int o = 64; o > 0; o >>= 1) {
    if ((int)threadIdx.x < o) red[threadIdx.x] += red[threadIdx.x + o];
    __syncthreads();
  }
  float inv = rsqrtf(red[0] / (float)H_ + 1e-5f);
  for (int j = threadIdx.x; j < H_; j += 128) xn[row * H_ + j] = xr[j] * inv * w[j];
}

// causal depthwise conv (K=4) over xs = xz[..., :E], then SiLU
__global__ void k_dwconv(const float* __restrict__ xz, const float* __restrict__ cw,
                         const float* __restrict__ cb, float* __restrict__ xc) {
  long long i = (long long)blockIdx.x * blockDim.x + threadIdx.x;
  if (i >= (long long)M_ * E_) return;
  int e = (int)(i % E_);
  long long bt = i / E_;
  int t = (int)(bt % T_);
  float acc = cb[e];
#pragma unroll
  for (int k = 0; k < KC_; ++k) {
    int ts = t - (KC_ - 1) + k;
    if (ts >= 0) acc += xz[(bt - t + ts) * (2 * E_) + e] * cw[e * KC_ + k];
  }
  xc[i] = acc / (1.0f + expf(-acc));
}

// sequential SSM scan; wave32 block: 2 channels x 16 states; includes D*u and z-gate
__global__ void k_scan(const float* __restrict__ delta, const float* __restrict__ dbc,
                       const float* __restrict__ xc, const float* __restrict__ xz,
                       const float* __restrict__ Alog, const float* __restrict__ Dp,
                       float* __restrict__ yg) {
  int lane = threadIdx.x;
  int n = lane & 15, half = lane >> 4;
  int b = blockIdx.x / (E_ / 2);
  int e = (blockIdx.x % (E_ / 2)) * 2 + half;
  float a_en = -expf(Alog[e * NS_ + n]);
  float dcoef = Dp[e];
  float h = 0.0f;
  for (int t = 0; t < T_; ++t) {
    long long bt = (long long)b * T_ + t;
    if (t + 16 < T_) {
      __builtin_prefetch(&delta[(bt + 16) * E_ + e], 0, 1);
      __builtin_prefetch(&xc[(bt + 16) * E_ + e], 0, 1);
    }
    float d = delta[bt * E_ + e];
    float u = xc[bt * E_ + e];
    float Bv = dbc[bt * 64 + R_ + n];
    float Cv = dbc[bt * 64 + R_ + NS_ + n];
    float dA = expf(d * a_en);
    h = dA * h + d * Bv * u;
    float p = h * Cv;
    p += __shfl_xor(p, 1);
    p += __shfl_xor(p, 2);
    p += __shfl_xor(p, 4);
    p += __shfl_xor(p, 8);
    if (n == 0) {
      float y = p + dcoef * u;
      float z = xz[bt * (2 * E_) + E_ + e];
      float sig = 1.0f / (1.0f + expf(-z));
      yg[bt * E_ + e] = y * (z * sig);
    }
  }
}

__global__ void k_copy(const float* __restrict__ a, float* __restrict__ o, long long n) {
  long long i = (long long)blockIdx.x * blockDim.x + threadIdx.x;
  if (i < n) o[i] = a[i];
}
__global__ void k_flip(const float* __restrict__ a, float* __restrict__ o) {
  long long i = (long long)blockIdx.x * blockDim.x + threadIdx.x;
  if (i >= (long long)M_ * H_) return;
  int h = (int)(i % H_);
  long long bt = i / H_;
  int t = (int)(bt % T_); long long b = bt / T_;
  o[i] = a[((b * T_) + (T_ - 1 - t)) * H_ + h];
}
__global__ void k_flipadd(const float* __restrict__ rev, float* __restrict__ xf) {
  long long i = (long long)blockIdx.x * blockDim.x + threadIdx.x;
  if (i >= (long long)M_ * H_) return;
  int h = (int)(i % H_);
  long long bt = i / H_;
  int t = (int)(bt % T_); long long b = bt / T_;
  xf[i] += rev[((b * T_) + (T_ - 1 - t)) * H_ + h];
}
__global__ void k_mean(const float* __restrict__ x, float* __restrict__ ctx) {
  int i = blockIdx.x * blockDim.x + threadIdx.x;
  if (i >= B_ * H_) return;
  int b = i / H_, h = i % H_;
  const float* p = x + (long long)b * T_ * H_ + h;
  float s = 0.0f;
  for (int t = 0; t < T_; ++t) s += p[(long long)t * H_];
  ctx[i] = s / (float)T_;
}
__global__ void k_combine(const float* __restrict__ f, const float* __restrict__ s,
                          const float* __restrict__ al, const float* __restrict__ be,
                          float* __restrict__ out) {
  int i = blockIdx.x * blockDim.x + threadIdx.x;
  if (i >= B_ * D_) return;
  out[i] = al[0] * f[i] + be[0] * s[i];
}

// ---------------------------------------------------------------------------
// Host-side helpers
// ---------------------------------------------------------------------------
static void gemm(const float* A, const __bf16* BT, const float* bias, float* C,
                 const float* zrow, int M, int N, int K, int lda, int ldc,
                 int shift, int Tper, int act, bool accum, hipStream_t s) {
  dim3 g(N / 64, (M + 15) / 16), blk(32);
  if (accum)          wmma_gemm<0, true ><<<g, blk, 0, s>>>(A, BT, bias, C, zrow, M, N, K, lda, ldc, shift, Tper);
  else if (act == 1)  wmma_gemm<1, false><<<g, blk, 0, s>>>(A, BT, bias, C, zrow, M, N, K, lda, ldc, shift, Tper);
  else if (act == 2)  wmma_gemm<2, false><<<g, blk, 0, s>>>(A, BT, bias, C, zrow, M, N, K, lda, ldc, shift, Tper);
  else                wmma_gemm<0, false><<<g, blk, 0, s>>>(A, BT, bias, C, zrow, M, N, K, lda, ldc, shift, Tper);
}

static void pack(const float* in, __bf16* out, long long off, long long sn,
                 long long sk, int N, int K, hipStream_t s) {
  long long tot = (long long)N * K;
  k_pack<<<(unsigned)((tot + 255) / 256), 256, 0, s>>>(in, out, off, sn, sk, N, K);
}

struct LayerW {
  const __bf16 *winT, *xprojT, *dtwT, *woutT;
  const float  *norm, *convw, *convb, *dtb, *Alog, *Dp;
};

static void mamba_layer(float* x, float* xn, float* xz, float* xc, float* dbc,
                        float* delta, float* yg, const float* zrow,
                        const LayerW& w, hipStream_t s) {
  k_rmsnorm<<<M_, 128, 0, s>>>(x, w.norm, xn);
  gemm(xn, w.winT, nullptr, xz, zrow, M_, 2 * E_, H_, H_, 2 * E_, 0, M_, 0, false, s);
  {
    long long tot = (long long)M_ * E_;
    k_dwconv<<<(unsigned)((tot + 255) / 256), 256, 0, s>>>(xz, w.convw, w.convb, xc);
  }
  gemm(xc, w.xprojT, nullptr, dbc, zrow, M_, R_ + 2 * NS_, E_, E_, 64, 0, M_, 0, false, s);
  gemm(dbc, w.dtwT, w.dtb, delta, zrow, M_, E_, R_, 64, E_, 0, M_, 2, false, s);
  k_scan<<<B_ * (E_ / 2), 32, 0, s>>>(delta, dbc, xc, xz, w.Alog, w.Dp, yg);
  gemm(yg, w.woutT, nullptr, x, zrow, M_, H_, E_, E_, H_, 0, M_, 0, true, s);
}

// ---------------------------------------------------------------------------
extern "C" void kernel_launch(void* const* d_in, const int* in_sizes, int n_in,
                              void* d_out, int out_size, void* d_ws, size_t ws_size,
                              hipStream_t stream) {
  const float* x_t     = (const float*)d_in[0];
  const int*   t_in    = (const int*)  d_in[1];
  const float* last    = (const float*)d_in[2];
  const float* fe      = (const float*)d_in[3];
  const float* W_in    = (const float*)d_in[4];
  const float* b_in    = (const float*)d_in[5];
  const float* fm_w1   = (const float*)d_in[6];
  const float* fm_b1   = (const float*)d_in[7];
  const float* fm_w2   = (const float*)d_in[8];
  const float* fm_b2   = (const float*)d_in[9];
  const float* c1w     = (const float*)d_in[10];
  const float* c1b     = (const float*)d_in[11];
  const float* c2w     = (const float*)d_in[12];
  const float* c2b     = (const float*)d_in[13];
  const float* te_w1   = (const float*)d_in[14];
  const float* te_b1   = (const float*)d_in[15];
  const float* te_w2   = (const float*)d_in[16];
  const float* te_b2   = (const float*)d_in[17];
  const float* m_norm  = (const float*)d_in[18];
  const float* m_inpr  = (const float*)d_in[19];
  const float* m_convw = (const float*)d_in[20];
  const float* m_convb = (const float*)d_in[21];
  const float* m_xproj = (const float*)d_in[22];
  const float* m_dtw   = (const float*)d_in[23];
  const float* m_dtb   = (const float*)d_in[24];
  const float* m_Alog  = (const float*)d_in[25];
  const float* m_D     = (const float*)d_in[26];
  const float* m_out   = (const float*)d_in[27];
  const float* fw1     = (const float*)d_in[28];
  const float* fb1     = (const float*)d_in[29];
  const float* fw2     = (const float*)d_in[30];
  const float* fb2     = (const float*)d_in[31];
  const float* sw1     = (const float*)d_in[32];
  const float* sb1     = (const float*)d_in[33];
  const float* sw2     = (const float*)d_in[34];
  const float* sb2     = (const float*)d_in[35];
  const float* alpha   = (const float*)d_in[36];
  const float* beta    = (const float*)d_in[37];

  char* base = (char*)d_ws;
  size_t off = 0;
  auto allocf = [&](long long n) -> float* {
    off = (off + 255) & ~(size_t)255;
    float* p = (float*)(base + off); off += (size_t)n * 4; return p;
  };
  auto allocb = [&](long long n) -> __bf16* {
    off = (off + 255) & ~(size_t)255;
    __bf16* p = (__bf16*)(base + off); off += (size_t)n * 2; return p;
  };

  // f32 activations
  float* zrow  = allocf(2048);            // shared zero row for masked A loads
  float* xin   = allocf((long long)M_ * D_);
  float* x     = allocf((long long)M_ * H_);
  float* tmpH  = allocf((long long)M_ * H_);
  float* xconv = allocf((long long)M_ * H_);
  float* xf    = allocf((long long)M_ * H_);
  float* xrev  = allocf((long long)M_ * H_);
  float* xn    = allocf((long long)M_ * H_);
  float* xz    = allocf((long long)M_ * 2 * E_);
  float* xc    = allocf((long long)M_ * E_);
  float* dbc   = allocf((long long)M_ * 64);
  float* delta = allocf((long long)M_ * E_);
  float* yg    = allocf((long long)M_ * E_);
  float* temb  = allocf(B_ * H_);
  float* ctx   = allocf(B_ * H_);
  float* hctx  = allocf(B_ * H_);
  float* fastb = allocf(B_ * D_);
  float* slowb = allocf(B_ * D_);

  // bf16 packed transposed weights: BT[n*K + k]
  __bf16* WinT  = allocb((long long)H_ * D_);
  __bf16* fw1T_ = allocb((long long)H_ * H_);
  __bf16* fw2T_ = allocb((long long)H_ * H_);
  __bf16* c1T   = allocb(3LL * H_ * H_);
  __bf16* c2T   = allocb(5LL * H_ * H_);
  __bf16* mWinT = allocb((long long)NL_ * 2 * E_ * H_);
  __bf16* mXpT  = allocb((long long)NL_ * 64 * E_);
  __bf16* mDtT  = allocb((long long)NL_ * E_ * R_);
  __bf16* mOutT = allocb((long long)NL_ * H_ * E_);
  __bf16* fa1T  = allocb((long long)H_ * H_);
  __bf16* fa2T  = allocb((long long)D_ * H_);
  __bf16* sl1T  = allocb((long long)H_ * H_);
  __bf16* sl2T  = allocb((long long)D_ * H_);

  k_zero<<<8, 256, 0, stream>>>(zrow, 2048);

  // ---- pack weights (N, K, strides into original row-major f32) ----
  pack(W_in,  WinT,  0, 1, H_, H_, D_, stream);                 // (D,H) -> (H,D)
  pack(fm_w1, fw1T_, 0, 1, H_, H_, H_, stream);
  pack(fm_w2, fw2T_, 0, 1, H_, H_, H_, stream);
  for (int k = 0; k < 3; ++k)  // conv1_w (O,I,3): slice k -> BT[o*H + c]
    pack(c1w, c1T + (long long)k * H_ * H_, k, (long long)H_ * 3, 3, H_, H_, stream);
  for (int k = 0; k < 5; ++k)
    pack(c2w, c2T + (long long)k * H_ * H_, k, (long long)H_ * 5, 5, H_, H_, stream);
  for (int i = 0; i < NL_; ++i) {
    pack(m_inpr, mWinT + (long long)i * 2 * E_ * H_, (long long)i * H_ * 2 * E_, 1, 2 * E_, 2 * E_, H_, stream);
    pack(m_xproj, mXpT + (long long)i * 64 * E_,     (long long)i * E_ * 64,     1, 64,      64,     E_, stream);
    pack(m_dtw,   mDtT + (long long)i * E_ * R_,     (long long)i * R_ * E_,     1, E_,      E_,     R_, stream);
    pack(m_out,   mOutT + (long long)i * H_ * E_,    (long long)i * E_ * H_,     1, H_,      H_,     E_, stream);
  }
  pack(fw1, fa1T, 0, 1, H_, H_, H_, stream);
  pack(fw2, fa2T, 0, 1, D_, D_, H_, stream);   // (H,D) -> (D,H)
  pack(sw1, sl1T, 0, 1, H_, H_, H_, stream);
  pack(sw2, sl2T, 0, 1, D_, D_, H_, stream);

  // ---- front end ----
  {
    long long tot = (long long)M_ * D_;
    k_prep<<<(unsigned)((tot + 255) / 256), 256, 0, stream>>>(x_t, last, fe, xin);
  }
  k_temb<<<B_, H_, 0, stream>>>(t_in, te_w1, te_b1, te_w2, te_b2, temb);
  gemm(xin, WinT, b_in, x, zrow, M_, H_, D_, D_, H_, 0, M_, 0, false, stream);
  {
    long long tot = (long long)M_ * H_;
    k_addtemb<<<(unsigned)((tot + 255) / 256), 256, 0, stream>>>(x, temb);
  }
  // FFN residual
  gemm(x, fw1T_, fm_b1, tmpH, zrow, M_, H_, H_, H_, H_, 0, M_, 1, false, stream);
  gemm(tmpH, fw2T_, fm_b2, x, zrow, M_, H_, H_, H_, H_, 0, M_, 0, true, stream);
  // conv1 (k=3, SAME -> shifts -1..1) + conv2 (k=5, shifts -2..2) as shifted GEMMs
  for (int k = 0; k < 3; ++k)
    gemm(x, c1T + (long long)k * H_ * H_, (k == 0) ? c1b : nullptr, xconv, zrow,
         M_, H_, H_, H_, H_, k - 1, T_, 0, (k != 0), stream);
  for (int k = 0; k < 5; ++k)
    gemm(x, c2T + (long long)k * H_ * H_, (k == 0) ? c2b : nullptr, xconv, zrow,
         M_, H_, H_, H_, H_, k - 2, T_, 0, true, stream);

  // ---- mamba stacks ----
  LayerW lw[NL_];
  for (int i = 0; i < NL_; ++i) {
    lw[i].winT   = mWinT + (long long)i * 2 * E_ * H_;
    lw[i].xprojT = mXpT  + (long long)i * 64 * E_;
    lw[i].dtwT   = mDtT  + (long long)i * E_ * R_;
    lw[i].woutT  = mOutT + (long long)i * H_ * E_;
    lw[i].norm   = m_norm  + (long long)i * H_;
    lw[i].convw  = m_convw + (long long)i * E_ * KC_;
    lw[i].convb  = m_convb + (long long)i * E_;
    lw[i].dtb    = m_dtb   + (long long)i * E_;
    lw[i].Alog   = m_Alog  + (long long)i * E_ * NS_;
    lw[i].Dp     = m_D     + (long long)i * E_;
  }
  long long nBTH = (long long)M_ * H_;
  k_copy<<<(unsigned)((nBTH + 255) / 256), 256, 0, stream>>>(xconv, xf, nBTH);
  for (int i = 0; i < NL_; ++i) mamba_layer(xf, xn, xz, xc, dbc, delta, yg, zrow, lw[i], stream);
  k_flip<<<(unsigned)((nBTH + 255) / 256), 256, 0, stream>>>(xconv, xrev);
  for (int i = 0; i < NL_; ++i) mamba_layer(xrev, xn, xz, xc, dbc, delta, yg, zrow, lw[i], stream);
  k_flipadd<<<(unsigned)((nBTH + 255) / 256), 256, 0, stream>>>(xrev, xf);

  // ---- heads ----
  k_mean<<<(B_ * H_ + 255) / 256, 256, 0, stream>>>(xf, ctx);
  gemm(ctx, fa1T, fb1, hctx, zrow, B_, H_, H_, H_, H_, 0, 16, 1, false, stream);
  gemm(hctx, fa2T, fb2, fastb, zrow, B_, D_, H_, H_, D_, 0, 16, 0, false, stream);
  gemm(ctx, sl1T, sb1, hctx, zrow, B_, H_, H_, H_, H_, 0, 16, 1, false, stream);
  gemm(hctx, sl2T, sb2, slowb, zrow, B_, D_, H_, H_, D_, 0, 16, 0, false, stream);
  k_combine<<<(B_ * D_ + 255) / 256, 256, 0, stream>>>(fastb, slowb, alpha, beta,
                                                       (float*)d_out);
}
